// GIN_38130719654022
// MI455X (gfx1250) — matrix operand order repaired and checked by reading, not compiled, not run on hardware
//
#include <hip/hip_runtime.h>
#include <hip/hip_bf16.h>

// ---------------- problem constants (from reference setup) ----------------
#define GG   32          // B*H
#define NN   1024        // nodes
#define FF   128         // F == NHID
#define KSEL 32768       // (N*N)//sqrt(N) top-k kept per matrix

typedef __attribute__((ext_vector_type(16))) __bf16 v16bf;
typedef __attribute__((ext_vector_type(8)))  float  v8f;

// Native f32 -> bf16 (RNE on gfx1250; backend emits v_cvt_pk_bf16_f32 pairs)
static __device__ __forceinline__ __bf16 tobf(float f) { return (__bf16)f; }

// ---- CDNA5 async global->LDS copy (ASYNCcnt-tracked), guarded by probe ----
#if defined(__AMDGCN__) && \
    __has_builtin(__builtin_amdgcn_global_load_async_to_lds_b128) && \
    __has_builtin(__builtin_amdgcn_s_wait_asynccnt)
#define HAVE_ASYNC_LDS 1
#else
#define HAVE_ASYNC_LDS 0
#endif

#if HAVE_ASYNC_LDS
#define ASG __attribute__((address_space(1)))
#define ASL __attribute__((address_space(3)))
// Builtin signature (from hipcc diagnostic): param0 = 'int __vector(4) __device__ *'
// i.e. pointer to 16-byte int vector in AS(1); param1 analogous in AS(3).
typedef int v4i_b128 __attribute__((vector_size(16)));
// ISA §10.2: generic LDS address truncated to 32 bits == LDS offset, so an
// integer round-trip gives a valid AS(3) pointer; AS(1) == generic numerically.
static __device__ __forceinline__ void async_cp16(const float* g, float* l) {
    __builtin_amdgcn_global_load_async_to_lds_b128(
        (ASG v4i_b128*)(unsigned long long)(size_t)g,
        (ASL v4i_b128*)(unsigned)(size_t)l, 0, 0);
}
#endif

// ---------------------------------------------------------------------------
// K0: exact radix-select of the k-th largest value per g.
// adj values are uniform [0,1) -> non-negative, so float bits are monotone.
// One 256-thread block per g; 4 passes of 8-bit digits, MSB first.
// ---------------------------------------------------------------------------
__global__ __launch_bounds__(256) void k_thresh(const float* __restrict__ adj,
                                                float* __restrict__ thr) {
    const int g = blockIdx.x;
    const float* a = adj + (size_t)g * NN * NN;
    __shared__ unsigned hist[256];
    __shared__ unsigned s_prefix;
    __shared__ unsigned s_kk;
    const int t = threadIdx.x;
    if (t == 0) { s_prefix = 0u; s_kk = (unsigned)KSEL; }

    for (int p = 0; p < 4; ++p) {
        hist[t] = 0u;
        __syncthreads();
        const int shift = 24 - 8 * p;
        const unsigned knownMask = (p == 0) ? 0u : (0xFFFFFFFFu << (shift + 8));
        const unsigned pref = s_prefix;
        for (int e = t; e < NN * NN; e += 256) {
            unsigned bits = __float_as_uint(a[e]);
            if ((bits & knownMask) == pref)
                atomicAdd(&hist[(bits >> shift) & 0xFFu], 1u);
        }
        __syncthreads();
        if (t == 0) {
            unsigned cum = 0, kk = s_kk;
            for (int b1 = 255; b1 >= 0; --b1) {
                unsigned c = hist[b1];
                if (cum + c >= kk) {
                    s_prefix = pref | ((unsigned)b1 << shift);
                    s_kk = kk - cum;
                    break;
                }
                cum += c;
            }
        }
        __syncthreads();
    }
    if (t == 0) thr[g] = __uint_as_float(s_prefix);
}

// ---------------------------------------------------------------------------
// K1: deg_i = 0.5*(rowsum_i(sp) + colsum_i(sp)); dinv = deg>0 ? deg^-1/2 : 0
// Block = 256 threads (64 x 4), covers 64 rows of one g; all loads coalesced.
// ---------------------------------------------------------------------------
__global__ __launch_bounds__(256) void k_deg(const float* __restrict__ adj,
                                             const float* __restrict__ thr,
                                             float* __restrict__ dinv) {
    const int g  = blockIdx.y;
    const int i0 = blockIdx.x * 64;
    const float th = thr[g];
    const float* a = adj + (size_t)g * NN * NN;
    __shared__ float rowAcc[64];
    __shared__ float colAcc[64];
    const int t  = threadIdx.x;
    const int tx = t & 63;
    const int ty = t >> 6;
    if (t < 64) { rowAcc[t] = 0.f; colAcc[t] = 0.f; }
    __syncthreads();

    float colLocal = 0.f;
    for (int s = 0; s < 16; ++s) {
        const int r = ty + 4 * s;
        float rowLocal = 0.f;
        for (int jt = 0; jt < 16; ++jt) {
            float v1 = a[(size_t)(i0 + r) * NN + jt * 64 + tx];  // row read
            rowLocal += (v1 >= th) ? v1 : 0.f;
            int j = s * 64 + jt * 4 + ty;                        // spans 0..1023
            float v2 = a[(size_t)j * NN + i0 + tx];              // column read (coalesced)
            colLocal += (v2 >= th) ? v2 : 0.f;
        }
        atomicAdd(&rowAcc[r], rowLocal);
    }
    atomicAdd(&colAcc[tx], colLocal);
    __syncthreads();
    if (t < 64) {
        float deg = 0.5f * (rowAcc[t] + colAcc[t]);
        dinv[g * NN + i0 + t] = (deg > 0.f) ? rsqrtf(deg) : 0.f;
    }
}

// ---------------------------------------------------------------------------
// K2: norm_adj[i][j] = dinv_i * dinv_j * 0.5*(sp(a_ij)+sp(a_ji))
// 16x16 LDS-transpose tiles; both global reads coalesced; writes coalesced.
// ---------------------------------------------------------------------------
__global__ __launch_bounds__(256) void k_norm(const float* __restrict__ adj,
                                              const float* __restrict__ thr,
                                              const float* __restrict__ dinv,
                                              float* __restrict__ adjOut) {
    const int g  = blockIdx.z;
    const int i0 = blockIdx.y * 16;
    const int j0 = blockIdx.x * 16;
    const int tx = threadIdx.x, ty = threadIdx.y;
    const float th = thr[g];
    const float* a = adj + (size_t)g * NN * NN;
    __shared__ float ldsT[16][17];

    float v2 = a[(size_t)(j0 + ty) * NN + i0 + tx];   // transpose source, coalesced
    ldsT[ty][tx] = (v2 >= th) ? v2 : 0.f;
    float v1 = a[(size_t)(i0 + ty) * NN + j0 + tx];
    float sp1 = (v1 >= th) ? v1 : 0.f;
    __syncthreads();
    float s   = 0.5f * (sp1 + ldsT[tx][ty]);          // sp(a[j0+tx][i0+ty])
    float val = s * dinv[g * NN + i0 + ty] * dinv[g * NN + j0 + tx];
    adjOut[(size_t)g * NN * NN + (size_t)(i0 + ty) * NN + j0 + tx] = val;
}

// ---------------------------------------------------------------------------
// K4: fused  h = gelu((x + norm_adj @ x) @ W + b)   via bf16 WMMA, f32 accum.
// Block = 256 threads (8 waves), covers 16 rows x 128 cols of one g.
// Wave w owns the 16x16 output tile at columns [16w,16w+16).
// Stage 1: agg tile, K=1024 in 32-chunks; A double-buffered in LDS via
//          global_load_async_to_lds_b128 (overlaps HBM stream with WMMA).
// Stage 2: second GEMM K=128 in 4 chunks, W converted on the fly (L2-hot).
// LDS rows padded to 36 floats: keeps B128 16B-aligned AND makes fragment
// reads bank-conflict-free (36*nrow mod 64 distinct for nrow=0..15).
// ---------------------------------------------------------------------------
__global__ __launch_bounds__(256) void k_fused(const float* __restrict__ x,
                                               const float* __restrict__ normadj,
                                               const float* __restrict__ W,
                                               const float* __restrict__ bias,
                                               float* __restrict__ hout) {
    const int g    = blockIdx.y;
    const int row0 = blockIdx.x * 16;
    const int t    = threadIdx.x;
    const int lane = t & 31;
    const int wave = t >> 5;       // 0..7
    const int col0 = wave * 16;    // output column tile base
    const int nrow = lane & 15;    // M (for A) or N (for B and C/D) within fragment
    const int khalf = lane >> 4;   // which K-half this lane holds

    __shared__ float ldsA[2][16][36];  // norm_adj chunk, double-buffered
    __shared__ float ldsH[16][129];    // h1 = x + agg (16 x 128, padded)

    const float* adjg = normadj + (size_t)g * NN * NN;
    const float* xg   = x       + (size_t)g * NN * FF;

    const int lr  = t >> 3;            // tile row, valid when t < 128
    const int lc4 = (t & 7) * 4;       // tile col (float4 granularity)

    // ---------- stage 1: agg = norm_adj @ x ----------
    v8f c = {};
#if HAVE_ASYNC_LDS
    if (t < 128)
        async_cp16(adjg + (size_t)(row0 + lr) * NN + lc4, &ldsA[0][lr][lc4]);
#endif
    for (int ch = 0; ch < NN / 32; ++ch) {
        const int kk = ch * 32;
#if HAVE_ASYNC_LDS
        const int cur = ch & 1;
        __builtin_amdgcn_s_wait_asynccnt(0);  // my async fill of 'cur' landed
        __syncthreads();                      // all fills visible; prev readers done
        if (ch + 1 < NN / 32 && t < 128)      // prefetch next chunk into other buffer
            async_cp16(adjg + (size_t)(row0 + lr) * NN + kk + 32 + lc4,
                       &ldsA[cur ^ 1][lr][lc4]);
#else
        const int cur = 0;
        __syncthreads();                      // previous chunk's readers done
        if (t < 128)
            *(float4*)&ldsA[0][lr][lc4] =
                *(const float4*)(adjg + (size_t)(row0 + lr) * NN + kk + lc4);
        __syncthreads();
#endif
        v16bf a, b;
        // ISA 7.12.2 16-bit A (16x32): lanes0-15 VGPR0-3 K=0..7, VGPR4-7 K=16..23;
        //                              lanes16-31 K=8..15 and K=24..31.
        // B (32x16): same K order as A (lanes0-15 K=0..7,16..23; lanes16-31 K=8..15,24..31).
#pragma unroll
        for (int j = 0; j < 16; ++j) {
            const int i = j >> 1, odd = j & 1;
            const int kA = (i < 4) ? (khalf * 8 + 2 * i + odd)
                                   : (16 + khalf * 8 + 2 * (i - 4) + odd);
            a[j] = tobf(ldsA[cur][nrow][kA]);
            const int kB = (j < 8) ? (khalf * 8 + j) : (16 + khalf * 8 + (j - 8));
            b[j] = tobf(xg[(size_t)(kk + kB) * FF + col0 + nrow]);
        }
        c = __builtin_amdgcn_wmma_f32_16x16x32_bf16(false, a, false, b,
                                                    (short)0, c, false, false);
    }

    // ---------- h1 = x + agg -> LDS (C/D layout: elem m -> row m+8*khalf) ----------
#pragma unroll
    for (int m = 0; m < 8; ++m) {
        const int r = m + khalf * 8;
        const int cc = col0 + nrow;
        ldsH[r][cc] = c[m] + xg[(size_t)(row0 + r) * FF + cc];
    }
    __syncthreads();

    // ---------- stage 2: gelu(h1 @ W + b) ----------
    v8f d = {};
#pragma unroll
    for (int ch = 0; ch < 4; ++ch) {
        const int kk0 = ch * 32;
        v16bf a, b;
#pragma unroll
        for (int j = 0; j < 16; ++j) {
            const int i = j >> 1, odd = j & 1;
            const int kA = kk0 + ((i < 4) ? (khalf * 8 + 2 * i + odd)
                                          : (16 + khalf * 8 + 2 * (i - 4) + odd));
            a[j] = tobf(ldsH[nrow][kA]);
            const int kB = kk0 + ((j < 8) ? (khalf * 8 + j) : (16 + khalf * 8 + (j - 8)));
            b[j] = tobf(W[kB * FF + col0 + nrow]);   // NHID == FF == 128
        }
        d = __builtin_amdgcn_wmma_f32_16x16x32_bf16(false, a, false, b,
                                                    (short)0, d, false, false);
    }
#pragma unroll
    for (int m = 0; m < 8; ++m) {
        const int r  = row0 + m + khalf * 8;
        const int cc = col0 + nrow;
        float v = d[m] + bias[cc];
        v = 0.5f * v * (1.0f + erff(v * 0.70710678118654752f));   // exact GELU
        hout[((size_t)g * NN + r) * FF + cc] = v;
    }
}

// ---------------------------------------------------------------------------
// Launcher
// ---------------------------------------------------------------------------
extern "C" void kernel_launch(void* const* d_in, const int* in_sizes, int n_in,
                              void* d_out, int out_size, void* d_ws, size_t ws_size,
                              hipStream_t stream) {
    (void)in_sizes; (void)n_in; (void)out_size; (void)ws_size;
    const float* x    = (const float*)d_in[0];   // [8,4,1024,128]
    const float* adj  = (const float*)d_in[1];   // [8,4,1024,1024]
    const float* W    = (const float*)d_in[2];   // [128,128]
    const float* bias = (const float*)d_in[3];   // [128]

    float* outp   = (float*)d_out;
    float* hOut   = outp;                                   // GG*NN*FF floats
    float* adjOut = outp + (size_t)GG * NN * FF;            // GG*NN*NN floats

    float* thr  = (float*)d_ws;                             // [GG]
    float* dinv = (float*)d_ws + 64;                        // [GG*NN]

    k_thresh<<<dim3(GG), dim3(256), 0, stream>>>(adj, thr);
    k_deg   <<<dim3(NN / 64, GG), dim3(256), 0, stream>>>(adj, thr, dinv);
    k_norm  <<<dim3(NN / 16, NN / 16, GG), dim3(16, 16), 0, stream>>>(adj, thr, dinv, adjOut);
    k_fused <<<dim3(NN / 16, GG), dim3(256), 0, stream>>>(x, adjOut, W, bias, hOut);
}
